// Attention_40286793236527
// MI455X (gfx1250) — 16-bit and 8-bit matrix operand order repaired and checked by reading, compiled, not run on hardware
//
#include <hip/hip_runtime.h>
#include <hip/hip_bf16.h>


typedef __attribute__((ext_vector_type(16))) _Float16 v16h;
typedef __attribute__((ext_vector_type(8)))  float    v8f;
typedef __attribute__((ext_vector_type(16))) int      v16i;

#define S_LEN 2048
#define MDIM  1024
#define HEADS 16
#define DHEAD 64
#define BSROWS 4096
#define NKB    8

union FragA8 { int4 q[4]; v16i v; };
union Frag16 { int4 q[2]; unsigned u[8]; v16h h; };

__device__ __forceinline__ unsigned char f32_to_e4m3(float x) {
  if (!(x == x)) return 0x7F;
  unsigned char sgn = (x < 0.0f) ? 0x80 : 0x00;
  float ax = fabsf(x);
  if (ax >= 464.0f) return sgn | 0x7E;
  if (ax < 0.0009765625f) return sgn;
  int e; float m = frexpf(ax, &e);
  int E = e + 6;
  if (E >= 1) {
    int mant = (int)(m * 16.0f + 0.5f) - 8;
    if (mant == 8) { mant = 0; ++E; }
    if (E > 15 || (E == 15 && mant > 6)) return sgn | 0x7E;
    return sgn | (unsigned char)((E << 3) | mant);
  }
  int q = (int)(ax * 512.0f + 0.5f);
  if (q > 7) return sgn | 0x08;
  return sgn | (unsigned char)q;
}

__global__ void pack_act_fp8(const float* __restrict__ X,
                             unsigned char* __restrict__ out) {
  long idx = (long)blockIdx.x * blockDim.x + threadIdx.x;
  int rt   = (int)(idx >> 14);
  int rem  = (int)(idx & 16383);
  int kb   = rem >> 11;
  int o    = rem & 2047;
  int lane = o >> 6;
  int t    = o & 63;
  int v = t >> 2, b = t & 3;
  int half = lane >> 4;
  int row  = (rt << 4) + (lane & 15);
  int vv   = v & 7;
  int K = (kb << 7) + ((v >> 3) << 6) + ((vv >> 1) << 4) + ((vv & 1) << 2)
        + (half << 3) + b;
  out[idx] = f32_to_e4m3(X[(long)row * MDIM + K]);
}

__global__ void pack_w_fp8(const float* __restrict__ W,
                           unsigned char* __restrict__ out) {
  int idx  = blockIdx.x * blockDim.x + threadIdx.x;
  int nt   = idx >> 14;
  int rem  = idx & 16383;
  int kb   = rem >> 11;
  int o    = rem & 2047;
  int lane = o >> 6;
  int t    = o & 63;
  int v = t >> 2, b = t & 3;
  int half = lane >> 4;
  int n = (nt << 4) + (lane & 15);
  int K = (kb << 7) + ((v >> 3) << 6) + (((v & 7) >> 1) << 4) + ((v & 1) << 2) + (half << 3) + b;
  int h = n >> 6, d = n & 63;
  out[idx] = f32_to_e4m3(W[((long)h * MDIM + K) * DHEAD + d]);
}

__global__ void transpose_wo(const float* __restrict__ WO,
                             _Float16* __restrict__ WoT) {
  int idx = blockIdx.x * blockDim.x + threadIdx.x;
  int n = idx >> 10, k = idx & 1023;
  WoT[idx] = (_Float16)WO[(long)k * MDIM + n];
}

__device__ __forceinline__ Frag16 load_a16(const _Float16* rowp, int k0, int half) {
  Frag16 f;
  f.q[0] = *(const int4*)(rowp + k0 + (half << 3));
  f.q[1] = *(const int4*)(rowp + k0 + 16 + (half << 3));
  return f;
}
__device__ __forceinline__ Frag16 load_b16(const _Float16* colp, int k0, int half) {
  Frag16 f;
  f.q[0] = *(const int4*)(colp + k0 + (half << 3));
  f.q[1] = *(const int4*)(colp + k0 + 16 + (half << 3));
  return f;
}

__global__ __launch_bounds__(32)
void qkv_proj_fp8(const unsigned char* __restrict__ qp,
                  const unsigned char* __restrict__ kp,
                  const unsigned char* __restrict__ vp,
                  const unsigned char* __restrict__ wq,
                  const unsigned char* __restrict__ wk,
                  const unsigned char* __restrict__ wv,
                  const float* __restrict__ bq,
                  const float* __restrict__ bk,
                  const float* __restrict__ bv,
                  _Float16* __restrict__ q16,
                  _Float16* __restrict__ k16,
                  _Float16* __restrict__ vT) {
  const int which = blockIdx.z;
  const unsigned char* A = which == 0 ? qp : which == 1 ? kp : vp;
  const unsigned char* W = which == 0 ? wq : which == 1 ? wk : wv;
  const float*      bias = which == 0 ? bq : which == 1 ? bk : bv;

  const int nb = blockIdx.x;
  const int mb = blockIdx.y;
  const int lane = threadIdx.x;
  const int half = lane >> 4, nn = lane & 15;

  v8f acc[2][4] = {{{}, {}, {}, {}}, {{}, {}, {}, {}}};

#pragma unroll
  for (int kb = 0; kb < NKB; ++kb) {
    FragA8 a[2], b[4];
#pragma unroll
    for (int i = 0; i < 2; ++i) {
      const int4* ap = (const int4*)(A + (((long)(mb * 2 + i) * NKB + kb) << 11))
                       + lane * 4;
#pragma unroll
      for (int t = 0; t < 4; ++t) a[i].q[t] = ap[t];
    }
#pragma unroll
    for (int j = 0; j < 4; ++j) {
      const int4* bp = (const int4*)(W + (((long)(nb * 4 + j) * NKB + kb) << 11))
                       + lane * 4;
#pragma unroll
      for (int t = 0; t < 4; ++t) b[j].q[t] = bp[t];
    }
#pragma unroll
    for (int i = 0; i < 2; ++i)
#pragma unroll
      for (int j = 0; j < 4; ++j)
        acc[i][j] = __builtin_amdgcn_wmma_f32_16x16x128_fp8_fp8(
            a[i].v, b[j].v, (short)0, acc[i][j], false, false);
  }

#pragma unroll
  for (int j = 0; j < 4; ++j) {
    const int col = (nb << 6) + (j << 4) + nn;
    const float bcol = bias[col];
    const int h = col >> 6, d = col & 63;
#pragma unroll
    for (int i = 0; i < 2; ++i)
#pragma unroll
      for (int r = 0; r < 8; ++r) {
        int row = (mb << 5) + (i << 4) + (half << 3) + r;
        int bb = row >> 11, s = row & (S_LEN - 1);
        float val = acc[i][j][r] + bcol;
        long bh = (long)(bb * HEADS + h);
        if (which < 2) {
          _Float16* dst = which == 0 ? q16 : k16;
          dst[(bh * S_LEN + s) * DHEAD + d] = (_Float16)val;
        } else {
          vT[(bh * DHEAD + d) * S_LEN + s] = (_Float16)val;
        }
      }
  }
}

__global__ __launch_bounds__(32)
void attn_fwd(const _Float16* __restrict__ q16,
              const _Float16* __restrict__ k16,
              const _Float16* __restrict__ vT,
              _Float16* __restrict__ z16) {
  const int qt = blockIdx.x;
  const int bh = blockIdx.y;
  const int lane = threadIdx.x;
  const int half = lane >> 4, nn = lane & 15;

  const _Float16* Q = q16 + (long)bh * S_LEN * DHEAD;
  const _Float16* K = k16 + (long)bh * S_LEN * DHEAD;
  const _Float16* V = vT  + (long)bh * DHEAD * S_LEN;

  Frag16 qa[2];
  const int qrow = (qt << 4) + nn;
#pragma unroll
  for (int dh = 0; dh < 2; ++dh)
    qa[dh] = load_a16(Q + (long)qrow * DHEAD, dh * 32, half);

  v8f o[4] = {{}, {}, {}, {}};
  float mrow[8], lrow[8];
#pragma unroll
  for (int r = 0; r < 8; ++r) { mrow[r] = -1e30f; lrow[r] = 0.0f; }

  __shared__ _Float16 P[16 * 40];

  const int nkb = ((qt << 4) + 16 + 31) >> 5;
  for (int kb = 0; kb < nkb; ++kb) {
    v8f sc[2];
#pragma unroll
    for (int j = 0; j < 2; ++j) {
      v8f s = {};
      const int key = kb * 32 + j * 16 + nn;
#pragma unroll
      for (int dh = 0; dh < 2; ++dh) {
        Frag16 kf = load_b16(K + (long)key * DHEAD, dh * 32, half);
        s = __builtin_amdgcn_wmma_f32_16x16x32_f16(
            false, qa[dh].h, false, kf.h, (short)0, s, false, false);
      }
#pragma unroll
      for (int r = 0; r < 8; ++r) {
        int qr = (qt << 4) + (half << 3) + r;
        float x = s[r] * 0.125f;
        if (key > qr) x = -1000.0f;
        s[r] = x;
      }
      sc[j] = s;
    }

#pragma unroll
    for (int r = 0; r < 8; ++r) {
      float mx = fmaxf(sc[0][r], sc[1][r]);
#pragma unroll
      for (int off = 8; off; off >>= 1) mx = fmaxf(mx, __shfl_xor(mx, off, 32));
      float nm = fmaxf(mrow[r], mx);
      float corr = __expf(mrow[r] - nm);
      mrow[r] = nm;
      float p0 = __expf(sc[0][r] - nm);
      float p1 = __expf(sc[1][r] - nm);
      sc[0][r] = p0; sc[1][r] = p1;
      float sum = p0 + p1;
#pragma unroll
      for (int off = 8; off; off >>= 1) sum += __shfl_xor(sum, off, 32);
      lrow[r] = lrow[r] * corr + sum;
      o[0][r] *= corr; o[1][r] *= corr; o[2][r] *= corr; o[3][r] *= corr;
    }

    __syncthreads();
#pragma unroll
    for (int r = 0; r < 8; ++r) {
      int row = (half << 3) + r;
      P[row * 40 + nn]      = (_Float16)sc[0][r];
      P[row * 40 + 16 + nn] = (_Float16)sc[1][r];
    }
    __syncthreads();
    Frag16 pa;
    pa.q[0] = *(const int4*)&P[nn * 40 + (half << 3)];
    pa.q[1] = *(const int4*)&P[nn * 40 + 16 + (half << 3)];

#pragma unroll
    for (int t4 = 0; t4 < 4; ++t4) {
      int col = (t4 << 4) + nn;
      Frag16 vf = load_b16(V + (long)col * S_LEN, kb * 32, half);
      o[t4] = __builtin_amdgcn_wmma_f32_16x16x32_f16(
          false, pa.h, false, vf.h, (short)0, o[t4], false, false);
    }
  }

  const int b = bh >> 4, h = bh & 15;
#pragma unroll
  for (int r = 0; r < 8; ++r) {
    float inv = 1.0f / lrow[r];
    int s = (qt << 4) + (half << 3) + r;
    long base = ((long)(b * S_LEN + s) << 10) + (h << 6) + nn;
#pragma unroll
    for (int t4 = 0; t4 < 4; ++t4)
      z16[base + (t4 << 4)] = (_Float16)(o[t4][r] * inv);
  }
}

__global__ __launch_bounds__(32)
void out_proj(const _Float16* __restrict__ z16,
              const _Float16* __restrict__ WoT,
              const float* __restrict__ bO,
              float* __restrict__ out) {
  const int nb = blockIdx.x;
  const int mb = blockIdx.y;
  const int lane = threadIdx.x;
  const int half = lane >> 4, nn = lane & 15;

  v8f acc[2][4] = {{{}, {}, {}, {}}, {{}, {}, {}, {}}};
  const long arow0 = (long)((mb << 5) + nn) * MDIM;
  const long arow1 = arow0 + (long)16 * MDIM;
  const long col0  = (long)((nb << 6) + nn) * MDIM;

#pragma unroll 2
  for (int kb = 0; kb < 32; ++kb) {
    Frag16 a[2], b[4];
    a[0] = load_a16(z16 + arow0, kb * 32, half);
    a[1] = load_a16(z16 + arow1, kb * 32, half);
#pragma unroll
    for (int j = 0; j < 4; ++j)
      b[j] = load_b16(WoT + col0 + (long)j * 16 * MDIM, kb * 32, half);
#pragma unroll
    for (int i = 0; i < 2; ++i)
#pragma unroll
      for (int j = 0; j < 4; ++j)
        acc[i][j] = __builtin_amdgcn_wmma_f32_16x16x32_f16(
            false, a[i].h, false, b[j].h, (short)0, acc[i][j], false, false);
  }

#pragma unroll
  for (int j = 0; j < 4; ++j) {
    const int col = (nb << 6) + (j << 4) + nn;
    const float bias = bO[col];
#pragma unroll
    for (int i = 0; i < 2; ++i)
#pragma unroll
      for (int r = 0; r < 8; ++r) {
        int row = (mb << 5) + (i << 4) + (half << 3) + r;
        out[(long)row * MDIM + col] = acc[i][j][r] + bias;
      }
  }
}

extern "C" void kernel_launch(void* const* d_in, const int* in_sizes, int n_in,
                              void* d_out, int out_size, void* d_ws, size_t ws_size,
                              hipStream_t stream) {
  const float* qin = (const float*)d_in[0];
  const float* kin = (const float*)d_in[1];
  const float* vin = (const float*)d_in[2];
  const float* WQ  = (const float*)d_in[3];
  const float* WK  = (const float*)d_in[4];
  const float* WV  = (const float*)d_in[5];
  const float* WO  = (const float*)d_in[6];
  const float* bQ  = (const float*)d_in[7];
  const float* bK  = (const float*)d_in[8];
  const float* bV  = (const float*)d_in[9];
  const float* bO  = (const float*)d_in[10];

  unsigned char* ws = (unsigned char*)d_ws;
  size_t o = 0;
  unsigned char* qpack = ws + o; o += (size_t)1 << 22;
  unsigned char* kpack = ws + o; o += (size_t)1 << 22;
  unsigned char* vpack = ws + o; o += (size_t)1 << 22;
  unsigned char* wqp   = ws + o; o += (size_t)1 << 20;
  unsigned char* wkp   = ws + o; o += (size_t)1 << 20;
  unsigned char* wvp   = ws + o; o += (size_t)1 << 20;
  _Float16* q16 = (_Float16*)(ws + o); o += (size_t)1 << 23;
  _Float16* k16 = (_Float16*)(ws + o); o += (size_t)1 << 23;
  _Float16* vT  = (_Float16*)(ws + o); o += (size_t)1 << 23;
  _Float16* z16 = (_Float16*)(ws + o); o += (size_t)1 << 23;
  _Float16* WoT = (_Float16*)(ws + o); o += (size_t)1 << 21;

  pack_act_fp8<<<16384, 256, 0, stream>>>(qin, qpack);
  pack_act_fp8<<<16384, 256, 0, stream>>>(kin, kpack);
  pack_act_fp8<<<16384, 256, 0, stream>>>(vin, vpack);
  pack_w_fp8<<<4096, 256, 0, stream>>>(WQ, wqp);
  pack_w_fp8<<<4096, 256, 0, stream>>>(WK, wkp);
  pack_w_fp8<<<4096, 256, 0, stream>>>(WV, wvp);
  transpose_wo<<<4096, 256, 0, stream>>>(WO, WoT);

  qkv_proj_fp8<<<dim3(16, 128, 3), 32, 0, stream>>>(
      qpack, kpack, vpack, wqp, wkp, wvp, bQ, bK, bV, q16, k16, vT);

  attn_fwd<<<dim3(128, 32), 32, 0, stream>>>(q16, k16, vT, z16);

  out_proj<<<dim3(16, 128), 32, 0, stream>>>(z16, WoT, bO, (float*)d_out);
}
